// DualWarping_51410758533574
// MI455X (gfx1250) — matrix operand order repaired and checked by reading, not compiled, not run on hardware
//
#include <hip/hip_runtime.h>
#include <hip/hip_bf16.h>

typedef __attribute__((ext_vector_type(16))) _Float16 v16h;
typedef __attribute__((ext_vector_type(8)))  _Float16 v8h;
typedef __attribute__((ext_vector_type(8)))  float    v8f;

#define B_    4
#define H_    512
#define W_    512
#define HF_   256
#define WF_   256
#define CF_   96
#define HW_   (H_ * W_)    // 262144
#define HWF_  (HF_ * WF_)  // 65536
#define CPAD  128
#define PW_   258          // padded spatial dim (1-px zero halo)
#define PHW_  (PW_ * PW_)  // 66564
#define NCHUNK 36          // K = 9 taps * 128 padded channels = 1152 = 36 * 32

// ---------------------------------------------------------------------------
// helpers
// ---------------------------------------------------------------------------
__device__ __forceinline__ void bilin_coord(float s, int limit, int& i0, int& i1, float& w) {
    float f = floorf(s);
    i0 = (int)f;
    i1 = min(i0 + 1, limit);
    w  = s - f;
}

__device__ __forceinline__ float bilin(const float* __restrict__ p, int W,
                                       int x0, int x1, int y0, int y1,
                                       float wx, float wy) {
    float v00 = p[y0 * W + x0];
    float v01 = p[y0 * W + x1];
    float v10 = p[y1 * W + x0];
    float v11 = p[y1 * W + x1];
    return v00 * (1.f - wx) * (1.f - wy) + v01 * wx * (1.f - wy) +
           v10 * (1.f - wx) * wy         + v11 * wx * wy;
}

// 2x downscale, jax.image.resize('linear', antialias=True): separable [1,3,3,1]/8
__device__ __forceinline__ float down4(const float* __restrict__ p, int oy, int ox) {
    const float wt0 = 0.125f, wt1 = 0.375f;
    float acc = 0.f;
    int yb = 2 * oy - 1, xb = 2 * ox - 1;
#pragma unroll
    for (int j = 0; j < 4; ++j) {
        int yy = min(max(yb + j, 0), H_ - 1);
        float wj = (j == 0 || j == 3) ? wt0 : wt1;
        float row = 0.f;
#pragma unroll
        for (int i = 0; i < 4; ++i) {
            int xx = min(max(xb + i, 0), W_ - 1);
            float wi = (i == 0 || i == 3) ? wt0 : wt1;
            row += wi * p[yy * W_ + xx];
        }
        acc += wj * row;
    }
    return acc;
}

// interior (unpadded y,x) -> element offset in padded NHWC-128 buffer
__device__ __forceinline__ size_t padded_off(int b, int y, int x) {
    return ((size_t)b * PHW_ + (size_t)(y + 1) * PW_ + (x + 1)) * CPAD;
}

// ---------------------------------------------------------------------------
// weight pre-swizzle into this file's A-fragment lane order (contiguous K halves).
// Layout: [octile(6)][chunk(36)][lane(32)][16 halfs].
// Lane l holds A rows m=l&15; half slot j -> K = chunk*32 + j + (l<16?0:16).
// K = tap*128 + ic.
// ---------------------------------------------------------------------------
__global__ void pack_weights_kernel(const float* __restrict__ w, _Float16* __restrict__ wp,
                                    int Cin) {
    int idx = blockIdx.x * blockDim.x + threadIdx.x;
    const int total = 6 * NCHUNK * 32 * 16;
    if (idx >= total) return;
    int j      = idx & 15;
    int lane   = (idx >> 4) & 31;
    int chunk  = (idx >> 9) % NCHUNK;
    int octile = (idx >> 9) / NCHUNK;
    int oc   = octile * 16 + (lane & 15);
    int koff = (lane < 16) ? 0 : 16;
    int k    = chunk * 32 + j + koff;
    int tap  = k >> 7;       // /128
    int ic   = k & 127;
    float v = 0.f;
    if (ic < Cin) {
        int kh = tap / 3, kw = tap % 3;
        v = w[((oc * Cin + ic) * 3 + kh) * 3 + kw];
    }
    wp[idx] = (_Float16)v;
}

// ---------------------------------------------------------------------------
// local_kernel_warp row pass: img [B,3,512,512] -> img_row [B,3,512,512]
// ---------------------------------------------------------------------------
__global__ void lkw_row_kernel(const float* __restrict__ img, const float* __restrict__ flow,
                               const float* __restrict__ kp, float* __restrict__ img_row) {
    int idx = blockIdx.x * blockDim.x + threadIdx.x;
    if (idx >= B_ * HW_) return;
    int b = idx >> 18;
    int pix = idx & (HW_ - 1);
    int y = pix >> 9, x = pix & 511;

    const float* kpb = kp + (size_t)b * 11 * HW_ + pix;
    float p0 = kpb[0], p1 = kpb[HW_], p2 = kpb[2 * HW_], p3 = kpb[3 * HW_], p4 = kpb[4 * HW_];
    float m = fmaxf(fmaxf(fmaxf(p0, p1), fmaxf(p2, p3)), p4);
    float e0 = expf(p0 - m), e1 = expf(p1 - m), e2 = expf(p2 - m), e3 = expf(p3 - m), e4 = expf(p4 - m);
    float inv = 1.f / (e0 + e1 + e2 + e3 + e4);
    float kw[5] = {e0 * inv, e1 * inv, e2 * inv, e3 * inv, e4 * inv};
    float aniso = 1.f / (1.f + expf(-kpb[10 * HW_])) + 0.5f;

    float fx = flow[(size_t)b * 2 * HW_ + pix];
    float fy = flow[(size_t)b * 2 * HW_ + HW_ + pix];
    float gx = 2.f * (x + 0.5f * fx) / (float)(W_ - 1) - 1.f;
    float gy = 2.f * (y + 0.5f * fy) / (float)(H_ - 1) - 1.f;
    float sy = fminf(fmaxf((gy + 1.f) * 0.5f * (H_ - 1), 0.f), (float)(H_ - 1));
    int y0, y1; float wy;
    bilin_coord(sy, H_ - 1, y0, y1, wy);

    const float* imb = img + (size_t)b * 3 * HW_;
    float acc[3] = {0.f, 0.f, 0.f};
#pragma unroll
    for (int t = 0; t < 5; ++t) {
        float off = (float)(t - 2) / aniso;
        float grx = gx + off * (2.f / (float)W_);
        float sx = fminf(fmaxf((grx + 1.f) * 0.5f * (W_ - 1), 0.f), (float)(W_ - 1));
        int x0, x1; float wx;
        bilin_coord(sx, W_ - 1, x0, x1, wx);
#pragma unroll
        for (int c = 0; c < 3; ++c)
            acc[c] += kw[t] * bilin(imb + (size_t)c * HW_, W_, x0, x1, y0, y1, wx, wy);
    }
#pragma unroll
    for (int c = 0; c < 3; ++c)
        img_row[(size_t)(b * 3 + c) * HW_ + pix] = acc[c];
}

// ---------------------------------------------------------------------------
// local_kernel_warp column pass: img_row -> warped_img (d_out region B)
// ---------------------------------------------------------------------------
__global__ void lkw_col_kernel(const float* __restrict__ img_row, const float* __restrict__ kp,
                               float* __restrict__ warped_img) {
    int idx = blockIdx.x * blockDim.x + threadIdx.x;
    if (idx >= B_ * HW_) return;
    int b = idx >> 18;
    int pix = idx & (HW_ - 1);
    int y = pix >> 9, x = pix & 511;

    const float* kpb = kp + (size_t)b * 11 * HW_ + pix;
    float p0 = kpb[5 * HW_], p1 = kpb[6 * HW_], p2 = kpb[7 * HW_], p3 = kpb[8 * HW_], p4 = kpb[9 * HW_];
    float m = fmaxf(fmaxf(fmaxf(p0, p1), fmaxf(p2, p3)), p4);
    float e0 = expf(p0 - m), e1 = expf(p1 - m), e2 = expf(p2 - m), e3 = expf(p3 - m), e4 = expf(p4 - m);
    float inv = 1.f / (e0 + e1 + e2 + e3 + e4);
    float kw[5] = {e0 * inv, e1 * inv, e2 * inv, e3 * inv, e4 * inv};
    float aniso = 1.f / (1.f + expf(-kpb[10 * HW_])) + 0.5f;

    float gx = 2.f * (float)x / (float)(W_ - 1) - 1.f;
    float gy = 2.f * (float)y / (float)(H_ - 1) - 1.f;
    float sx = fminf(fmaxf((gx + 1.f) * 0.5f * (W_ - 1), 0.f), (float)(W_ - 1));
    int x0, x1; float wx;
    bilin_coord(sx, W_ - 1, x0, x1, wx);

    const float* imb = img_row + (size_t)b * 3 * HW_;
    float acc[3] = {0.f, 0.f, 0.f};
#pragma unroll
    for (int t = 0; t < 5; ++t) {
        float off = (float)(t - 2) * aniso;
        float gcy = gy + off * (2.f / (float)H_);
        float sy = fminf(fmaxf((gcy + 1.f) * 0.5f * (H_ - 1), 0.f), (float)(H_ - 1));
        int y0, y1; float wy;
        bilin_coord(sy, H_ - 1, y0, y1, wy);
#pragma unroll
        for (int c = 0; c < 3; ++c)
            acc[c] += kw[t] * bilin(imb + (size_t)c * HW_, W_, x0, x1, y0, y1, wx, wy);
    }
#pragma unroll
    for (int c = 0; c < 3; ++c)
        warped_img[(size_t)(b * 3 + c) * HW_ + pix] = acc[c];
}

// ---------------------------------------------------------------------------
// backward_warp of features with 2x-downscaled flow.
// Writes f32 warped_feat to d_out region A and f16 copy to padded NHWC-128 xbuf ch 0..95.
// ---------------------------------------------------------------------------
__global__ void warp_feat_kernel(const float* __restrict__ feat, const float* __restrict__ flow,
                                 float* __restrict__ out_feat, _Float16* __restrict__ xbuf) {
    int idx = blockIdx.x * blockDim.x + threadIdx.x;
    if (idx >= B_ * HWF_) return;
    int b = idx >> 16;
    int pix = idx & (HWF_ - 1);
    int py = pix >> 8, px = pix & 255;

    const float* fx_p = flow + (size_t)b * 2 * HW_;
    const float* fy_p = fx_p + HW_;
    float fsx = down4(fx_p, py, px) * 0.5f;   // resize(linear, antialias) * (Hf/H)
    float fsy = down4(fy_p, py, px) * 0.5f;

    float vx = (float)px + 0.5f * fsx;
    float vy = (float)py + 0.5f * fsy;
    float gx = 2.f * vx / (float)(WF_ - 1) - 1.f;
    float gy = 2.f * vy / (float)(HF_ - 1) - 1.f;
    float valid = (fabsf(gx) <= 1.f && fabsf(gy) <= 1.f) ? 1.f : 0.f;
    float sx = fminf(fmaxf((gx + 1.f) * 0.5f * (WF_ - 1), 0.f), (float)(WF_ - 1));
    float sy = fminf(fmaxf((gy + 1.f) * 0.5f * (HF_ - 1), 0.f), (float)(HF_ - 1));
    int x0, x1, y0, y1; float wx, wy;
    bilin_coord(sx, WF_ - 1, x0, x1, wx);
    bilin_coord(sy, HF_ - 1, y0, y1, wy);

    const float* fb = feat + (size_t)b * CF_ * HWF_;
    _Float16* xb = xbuf + padded_off(b, py, px);
    for (int c = 0; c < CF_; ++c) {
        float v = valid * bilin(fb + (size_t)c * HWF_, WF_, x0, x1, y0, y1, wx, wy);
        out_feat[(size_t)(b * CF_ + c) * HWF_ + pix] = v;
        xb[c] = (_Float16)v;
    }
}

// ---------------------------------------------------------------------------
// downscale warped_img -> xbuf interior ch 96..98 (f16), zero ch 99..127
// ---------------------------------------------------------------------------
__global__ void down_img_kernel(const float* __restrict__ warped_img, _Float16* __restrict__ xbuf) {
    int idx = blockIdx.x * blockDim.x + threadIdx.x;
    if (idx >= B_ * HWF_) return;
    int b = idx >> 16;
    int pix = idx & (HWF_ - 1);
    int py = pix >> 8, px = pix & 255;
    _Float16* xb = xbuf + padded_off(b, py, px);
#pragma unroll
    for (int c = 0; c < 3; ++c)
        xb[CF_ + c] = (_Float16)down4(warped_img + (size_t)(b * 3 + c) * HW_, py, px);
    for (int c = CF_ + 3; c < CPAD; ++c)
        xb[c] = (_Float16)0.f;
}

// zero padded channels 96..127 of y1 interior (conv1 only writes 0..95)
__global__ void zero_y1_pad_kernel(_Float16* __restrict__ y1) {
    int idx = blockIdx.x * blockDim.x + threadIdx.x;
    if (idx >= B_ * HWF_) return;
    int b = idx >> 16;
    int pix = idx & (HWF_ - 1);
    _Float16* yb = y1 + padded_off(b, pix >> 8, pix & 255) + CF_;
    v8h z = {};
#pragma unroll
    for (int i = 0; i < 4; ++i)
        *(v8h*)(yb + i * 8) = z;
}

// zero all 128 channels on the 1-px spatial halo of a padded buffer
__global__ void zero_halo_kernel(_Float16* __restrict__ buf) {
    int idx = blockIdx.x * blockDim.x + threadIdx.x;
    if (idx >= B_ * PHW_) return;
    int pix = idx % PHW_;
    int r = pix / PW_, c = pix % PW_;
    if (r != 0 && r != PW_ - 1 && c != 0 && c != PW_ - 1) return;
    _Float16* p = buf + (size_t)idx * CPAD;
    v8h z = {};
#pragma unroll
    for (int i = 0; i < 16; ++i)
        *(v8h*)(p + i * 8) = z;
}

// ---------------------------------------------------------------------------
// implicit-GEMM 3x3 conv via v_wmma_f32_16x16x32_f16 on spatially-padded input.
// Wave tile: 16 oc x 32 pixels (two accumulators share each A fragment).
// Fully unrolled K loop: every tap/channel offset folds to an immediate, and
// the scheduler sees the whole load/WMMA DAG so it can hoist load clauses and
// use partial s_wait_loadcnt instead of a loop-carried full drain.
// RELU_TO_HALF=true : +bias, ReLU, store f16 padded NHWC-128 (conv1 -> y1)
// RELU_TO_HALF=false: +bias, accumulate into f32 NCHW d_out (conv2)
// ---------------------------------------------------------------------------
template <bool RELU_TO_HALF>
__global__ void conv3x3_wmma_kernel(const _Float16* __restrict__ xin,   // padded NHWC-128 f16
                                    const _Float16* __restrict__ wp,    // swizzled weights
                                    const float* __restrict__ bias,
                                    _Float16* __restrict__ yhalf,       // padded NHWC-128 f16
                                    float* __restrict__ yfloat) {       // NCHW f32
    const int bz     = blockIdx.z;
    const int octile = blockIdx.y;
    const int y      = blockIdx.x;                // one block per image row
    const int wv     = threadIdx.x >> 5;
    const int lane   = threadIdx.x & 31;
    const int n      = lane & 15;
    const int x0     = wv * 32 + n;
    const int x1     = x0 + 16;

    // loop-invariant bases; all per-chunk deltas below are compile-time consts
    const _Float16* wpt = wp + (size_t)octile * (NCHUNK * 32 * 16) + lane * 16;
    const int hi = (lane < 16) ? 0 : 16;          // B-fragment K-half select
    const _Float16* xrow0 = xin + (size_t)bz * PHW_ * CPAD + (size_t)y * (PW_ * CPAD) + hi;
    const _Float16* xpix0 = xrow0 + (size_t)x0 * CPAD;
    const _Float16* xpix1 = xrow0 + (size_t)x1 * CPAD;

    v8f acc0 = {}, acc1 = {};
#pragma unroll
    for (int chunk = 0; chunk < NCHUNK; ++chunk) {
        const int tap = chunk >> 2;               // constant after unroll
        const int icb = (chunk & 3) << 5;
        const int dy  = tap / 3, dx = tap % 3;    // padded coords: (y+dy, x+dx)
        const int off = (dy * PW_ + dx) * CPAD + icb;
        v16h a  = *(const v16h*)(wpt + chunk * (32 * 16));
        v16h b0 = *(const v16h*)(xpix0 + off);
        v16h b1 = *(const v16h*)(xpix1 + off);
        acc0 = __builtin_amdgcn_wmma_f32_16x16x32_f16(
            false, a, false, b0, (short)0, acc0, false, false);
        acc1 = __builtin_amdgcn_wmma_f32_16x16x32_f16(
            false, a, false, b1, (short)0, acc1, false, false);
    }

    const int mbase = (lane < 16) ? 0 : 8;
    const int ocb   = octile * 16 + mbase;
    if (RELU_TO_HALF) {
        v8h o0, o1;
#pragma unroll
        for (int r = 0; r < 8; ++r) {
            float bi = bias[ocb + r];
            float v0 = acc0[r] + bi;
            float v1 = acc1[r] + bi;
            o0[r] = (_Float16)(v0 > 0.f ? v0 : 0.f);
            o1[r] = (_Float16)(v1 > 0.f ? v1 : 0.f);
        }
        *(v8h*)(yhalf + padded_off(bz, y, x0) + ocb) = o0;
        *(v8h*)(yhalf + padded_off(bz, y, x1) + ocb) = o1;
    } else {
        float* ob = yfloat + (size_t)bz * CF_ * HWF_ + (size_t)y * WF_;
#pragma unroll
        for (int r = 0; r < 8; ++r) {
            float bi = bias[ocb + r];
            float* oc_p = ob + (size_t)(ocb + r) * HWF_;
            oc_p[x0] += acc0[r] + bi;
            oc_p[x1] += acc1[r] + bi;
        }
    }
}

// ---------------------------------------------------------------------------
extern "C" void kernel_launch(void* const* d_in, const int* in_sizes, int n_in,
                              void* d_out, int out_size, void* d_ws, size_t ws_size,
                              hipStream_t stream) {
    const float* img  = (const float*)d_in[0];
    const float* feat = (const float*)d_in[1];
    const float* flow = (const float*)d_in[2];
    const float* kp   = (const float*)d_in[3];
    const float* w1   = (const float*)d_in[4];
    const float* b1   = (const float*)d_in[5];
    const float* w2   = (const float*)d_in[6];
    const float* b2   = (const float*)d_in[7];

    float* out      = (float*)d_out;
    float* out_feat = out;                                 // [4,96,256,256]
    float* out_wimg = out + (size_t)B_ * CF_ * HWF_;       // [4,3,512,512]

    char* ws = (char*)d_ws;
    const size_t IMG_ROW_B = (size_t)B_ * 3 * HW_ * 4;       // 12,582,912
    const size_t PBUF_B    = (size_t)B_ * PHW_ * CPAD * 2;   // 68,161,536
    float*     img_row = (float*)ws;
    _Float16*  xbuf    = (_Float16*)(ws + IMG_ROW_B);
    _Float16*  y1      = (_Float16*)(ws + IMG_ROW_B + PBUF_B);
    _Float16*  w1p     = (_Float16*)(ws + IMG_ROW_B + 2 * PBUF_B);
    _Float16*  w2p     = w1p + 6 * NCHUNK * 32 * 16;

    const int NT = 256;
    const int wtotal = 6 * NCHUNK * 32 * 16;               // 110592
    pack_weights_kernel<<<(wtotal + NT - 1) / NT, NT, 0, stream>>>(w1, w1p, CF_ + 3);
    pack_weights_kernel<<<(wtotal + NT - 1) / NT, NT, 0, stream>>>(w2, w2p, CF_);

    const int nhalo = B_ * PHW_;                           // 266,256
    zero_halo_kernel<<<(nhalo + NT - 1) / NT, NT, 0, stream>>>(xbuf);
    zero_halo_kernel<<<(nhalo + NT - 1) / NT, NT, 0, stream>>>(y1);

    const int nfull = B_ * HW_;                            // 1,048,576
    lkw_row_kernel<<<nfull / NT, NT, 0, stream>>>(img, flow, kp, img_row);
    lkw_col_kernel<<<nfull / NT, NT, 0, stream>>>(img_row, kp, out_wimg);

    const int nhalf = B_ * HWF_;                           // 262,144
    warp_feat_kernel<<<nhalf / NT, NT, 0, stream>>>(feat, flow, out_feat, xbuf);
    down_img_kernel<<<nhalf / NT, NT, 0, stream>>>(out_wimg, xbuf);
    zero_y1_pad_kernel<<<nhalf / NT, NT, 0, stream>>>(y1);

    dim3 cgrid(HF_, 6, B_);                                // rows x oc-tiles x batch
    conv3x3_wmma_kernel<true ><<<cgrid, NT, 0, stream>>>(xbuf, w1p, b1, y1, nullptr);
    conv3x3_wmma_kernel<false><<<cgrid, NT, 0, stream>>>(y1,   w2p, b2, nullptr, out_feat);
}
